// RecommendationModel_13804024889530
// MI455X (gfx1250) — matrix operand order repaired and checked by reading, not compiled, not run on hardware
//
#include <hip/hip_runtime.h>
#include <cstdint>
#include <cstddef>

#define HN      128
#define NUM_U   50000
#define NUM_REC 100000
#define NUM_ING 20000
#define NEDGE_IT 800000
#define NEDGE_UI 1000000
#define WSLOT   16384            // f16 elements per packed weight slot (128x128)

typedef __attribute__((ext_vector_type(16))) _Float16 v16h;
typedef __attribute__((ext_vector_type(4)))  _Float16 h4;
typedef __attribute__((ext_vector_type(8)))  float    v8f;

__device__ __forceinline__ float wave_sum(float v) {
#pragma unroll
  for (int o = 16; o; o >>= 1) v += __shfl_xor(v, o, 32);
  return v;
}

// ---------------------------------------------------------------------------
// Pack W[K,128] (f32, row-major) into f16 WMMA B-fragment order:
//   Wp[((s*8 + t)*32 + l)*16 + j] = W[(s*32 + (l>>4)*16 + j)*128 + t*16 + (l&15)]
// so each (kstep s, col-tile t, lane l) owns 16 contiguous f16 (32 bytes).
// Rows k >= K are zero-filled (handles K=20 padding).
// ---------------------------------------------------------------------------
__global__ void pack_w(const float* __restrict__ W, _Float16* __restrict__ Wp, int K) {
  int Kp = (K + 31) & ~31;
  int idx = blockIdx.x * 256 + threadIdx.x;
  if (idx >= Kp * HN) return;
  int j = idx & 15;
  int l = (idx >> 4) & 31;
  int t = (idx >> 9) & 7;
  int s = idx >> 12;
  int n = t * 16 + (l & 15);
  int k = s * 32 + (l >> 4) * 16 + j;
  float v = (k < K) ? W[(size_t)k * HN + n] : 0.0f;
  Wp[idx] = (_Float16)v;
}

// ---------------------------------------------------------------------------
// Y[M,128] = act(X[M,K] @ W[K,128] + b) via v_wmma_f32_16x16x32_f16.
// Block = 256 threads (8 wave32); block owns a 16-row stripe (M % 16 == 0),
// wave w owns the 16-column tile n0 = 16*w. X tile staged to LDS as f16.
// B-fragments come pre-packed from pack_w (one unguarded 32B load / K-step).
// MODE 0: store.  MODE 1: relu + store.  MODE 2: tanh + atomic column-sum.
// ---------------------------------------------------------------------------
template <int MODE>
__global__ __launch_bounds__(256)
void gemm_wmma(const float* __restrict__ X, const _Float16* __restrict__ Wp,
               const float* __restrict__ bias, float* __restrict__ Y,
               float* __restrict__ colsum, int M, int K) {
  __shared__ _Float16 As[16 * HN];             // 16 x Kp (Kp <= 128)
  const int Kp   = (K + 31) & ~31;
  const int tid  = threadIdx.x;
  const int wave = tid >> 5;
  const int lane = tid & 31;
  const int m0   = blockIdx.x * 16;

  // Stage X tile (f32 -> f16), float4 loads + 8B LDS stores. K % 4 == 0, so a
  // 4-wide group is either fully valid (c < K) or fully in the zero pad.
  const int ngrp = 16 * Kp / 4;
  for (int g = tid; g < ngrp; g += 256) {
    int r = (g * 4) / Kp, c = (g * 4) % Kp;
    h4 hv;
    if (c < K) {
      float4 xv = *reinterpret_cast<const float4*>(X + (size_t)(m0 + r) * K + c);
      hv[0] = (_Float16)xv.x; hv[1] = (_Float16)xv.y;
      hv[2] = (_Float16)xv.z; hv[3] = (_Float16)xv.w;
    } else {
      hv[0] = hv[1] = hv[2] = hv[3] = (_Float16)0.0f;
    }
    *reinterpret_cast<h4*>(As + r * Kp + c) = hv;
  }
  __syncthreads();

  const int nn   = lane & 15;
  const int half = lane >> 4;                  // 0: lanes 0-15, 1: lanes 16-31
  v8f acc = {};

  for (int kk = 0; kk < Kp; kk += 32) {
    // A fragment (16x32 f16): one contiguous 32B run per lane (k = 16*half + s).
    v16h a;
    const int kh = half * 8;
#pragma unroll
    for (int v = 0; v < 8; ++v) {
      int kbase = half * 16 + 2 * v;
      a[2 * v]     = As[nn * Kp + kk + kbase];
      a[2 * v + 1] = As[nn * Kp + kk + kbase + 1];
    }
    // B fragment: one 32-byte pre-packed load.
    v16h b = *reinterpret_cast<const v16h*>(
        Wp + ((((kk >> 5) * 8 + wave) << 5) + lane) * 16);
    acc = __builtin_amdgcn_wmma_f32_16x16x32_f16(false, a, false, b,
                                                 (short)0, acc, false, false);
  }

  const int   n0    = wave * 16;
  const float bv    = bias[n0 + nn];
  const int   mbase = half * 8;                 // C/D: VGPR v -> M=v / M=v+8
  if constexpr (MODE == 2) {
    float part = 0.0f;
#pragma unroll
    for (int v = 0; v < 8; ++v) part += tanhf(acc[v] + bv);
    atomicAdd(&colsum[n0 + nn], part);
  } else {
#pragma unroll
    for (int v = 0; v < 8; ++v) {
      float y = acc[v] + bv;
      if constexpr (MODE == 1) y = y > 0.0f ? y : 0.0f;
      Y[(size_t)(m0 + mbase + v) * HN + n0 + nn] = y;
    }
  }
}

// --------------------------- row-wise helpers ------------------------------
// One wave per row (8 rows per 256-thread block).

__global__ void gather_renorm(const float* __restrict__ emb, const int* __restrict__ ids,
                              float* __restrict__ out, int n) {
  int wv = threadIdx.x >> 5, ln = threadIdx.x & 31;
  int row = blockIdx.x * 8 + wv;
  if (row >= n) return;
  const float* src = emb + (size_t)ids[row] * HN;
  float v[4], ss = 0.0f;
#pragma unroll
  for (int j = 0; j < 4; ++j) { v[j] = src[ln + 32 * j]; ss += v[j] * v[j]; }
  ss = wave_sum(ss);
  float sc = fminf(1.0f, 1.0f / fmaxf(sqrtf(ss), 1e-7f));
#pragma unroll
  for (int j = 0; j < 4; ++j) out[(size_t)row * HN + ln + 32 * j] = v[j] * sc;
}

__global__ void l2_normalize(float* __restrict__ x, int n) {
  int wv = threadIdx.x >> 5, ln = threadIdx.x & 31;
  int row = blockIdx.x * 8 + wv;
  if (row >= n) return;
  float* xr = x + (size_t)row * HN;
  float v[4], ss = 0.0f;
#pragma unroll
  for (int j = 0; j < 4; ++j) { v[j] = xr[ln + 32 * j]; ss += v[j] * v[j]; }
  ss = wave_sum(ss);
  float sc = 1.0f / fmaxf(sqrtf(ss), 1e-12f);
#pragma unroll
  for (int j = 0; j < 4; ++j) xr[ln + 32 * j] = v[j] * sc;
}

__global__ void row_dot(const float* __restrict__ x, const float* __restrict__ a,
                        float* __restrict__ out, int n) {
  int wv = threadIdx.x >> 5, ln = threadIdx.x & 31;
  int row = blockIdx.x * 8 + wv;
  if (row >= n) return;
  const float* xr = x + (size_t)row * HN;
  float p = 0.0f;
#pragma unroll
  for (int j = 0; j < 4; ++j) p += xr[ln + 32 * j] * a[ln + 32 * j];
  p = wave_sum(p);
  if (ln == 0) out[row] = p;
}

// --------------------------- edge attention --------------------------------
// Monotonic float->uint key so atomicMax(uint) implements float segment-max.
__device__ __forceinline__ unsigned fkey(float x) {
  unsigned u = __float_as_uint(x);
  return (u & 0x80000000u) ? ~u : (u | 0x80000000u);
}
__device__ __forceinline__ float funkey(unsigned k) {
  unsigned u = (k & 0x80000000u) ? (k ^ 0x80000000u) : ~k;
  return __uint_as_float(u);
}

__global__ void edge_alpha_kernel(const float* __restrict__ as, const float* __restrict__ ad,
                                  const int* __restrict__ src, const int* __restrict__ dst,
                                  float* __restrict__ alpha, unsigned* __restrict__ mkey, int E) {
  int e = blockIdx.x * blockDim.x + threadIdx.x;
  if (e >= E) return;
  float x = as[src[e]] + ad[dst[e]];
  float al = x > 0.0f ? x : 0.2f * x;          // leaky_relu(0.2)
  alpha[e] = al;
  atomicMax(&mkey[dst[e]], fkey(al));
}

__global__ void edge_exp_kernel(float* __restrict__ alpha, const unsigned* __restrict__ mkey,
                                const int* __restrict__ dst, float* __restrict__ segs, int E) {
  int e = blockIdx.x * blockDim.x + threadIdx.x;
  if (e >= E) return;
  int d = dst[e];
  float ee = expf(alpha[e] - funkey(mkey[d]));
  alpha[e] = ee;                                // in-place: alpha -> exp term
  atomicAdd(&segs[d], ee);
}

__global__ void edge_scatter_kernel(const float* __restrict__ hs, const float* __restrict__ ev,
                                    const float* __restrict__ segs, const int* __restrict__ src,
                                    const int* __restrict__ dst, float* __restrict__ out, int E) {
  int e = blockIdx.x;
  if (e >= E) return;
  int d = dst[e];
  float w = ev[e] / (segs[d] + 1e-16f);
  int f = threadIdx.x;                          // 128 threads = 1 column each
  atomicAdd(&out[(size_t)d * HN + f], hs[(size_t)src[e] * HN + f] * w);
}

__global__ void relu_kernel(float* __restrict__ x, size_t n) {
  size_t i = (size_t)blockIdx.x * blockDim.x + threadIdx.x;
  if (i < n) { float v = x[i]; x[i] = v > 0.0f ? v : 0.0f; }
}

// --------------------------- semantic attention ----------------------------
__global__ void group_softmax(const float* __restrict__ colsum, const float* __restrict__ q,
                              float* __restrict__ attn, float invN) {
  __shared__ float sc[4];
  int wv = threadIdx.x >> 5, ln = threadIdx.x & 31;   // 4 waves, one per type
  float p = 0.0f;
#pragma unroll
  for (int j = 0; j < 4; ++j) {
    int f = ln + 32 * j;
    p += q[f] * colsum[wv * HN + f];
  }
  p = wave_sum(p);
  if (ln == 0) sc[wv] = p * invN;
  __syncthreads();
  if (threadIdx.x == 0) {
    float m = fmaxf(fmaxf(sc[0], sc[1]), fmaxf(sc[2], sc[3]));
    float e0 = expf(sc[0] - m), e1 = expf(sc[1] - m);
    float e2 = expf(sc[2] - m), e3 = expf(sc[3] - m);
    float s = e0 + e1 + e2 + e3;
    attn[0] = e0 / s; attn[1] = e1 / s; attn[2] = e2 / s; attn[3] = e3 / s;
  }
}

__global__ void combine4(const float* __restrict__ a, const float* __restrict__ b,
                         const float* __restrict__ c, const float* __restrict__ d,
                         const float* __restrict__ attn, float* __restrict__ out, size_t n) {
  size_t i = (size_t)blockIdx.x * blockDim.x + threadIdx.x;
  if (i < n) out[i] = attn[0] * a[i] + attn[1] * b[i] + attn[2] * c[i] + attn[3] * d[i];
}

// --------------------------- batchnorm -------------------------------------
__global__ void col_stats(const float* __restrict__ x, int M,
                          float* __restrict__ sum, float* __restrict__ sumsq) {
  int f = threadIdx.x;                          // 128 threads, 1 column each
  float s = 0.0f, q = 0.0f;
  for (int r = blockIdx.x; r < M; r += gridDim.x) {
    float v = x[(size_t)r * HN + f];
    s += v; q += v * v;
  }
  atomicAdd(&sum[f], s);
  atomicAdd(&sumsq[f], q);
}

__global__ void bn_apply(const float* __restrict__ x, const float* __restrict__ sum,
                         const float* __restrict__ sumsq, const float* __restrict__ g,
                         const float* __restrict__ b, float* __restrict__ out, int M) {
  int f = threadIdx.x;
  float mu = sum[f] / (float)M;
  float var = sumsq[f] / (float)M - mu * mu;    // biased var, matches x.var(0)
  float rs = rsqrtf(var + 1e-5f);
  float gg = g[f], bb = b[f];
  for (int r = blockIdx.x; r < M; r += gridDim.x)
    out[(size_t)r * HN + f] = (x[(size_t)r * HN + f] - mu) * rs * gg + bb;
}

// --------------------------- host-side helpers -----------------------------
static void launch_gemm(const float* X, const _Float16* Wp, const float* bias, float* Y,
                        float* colsum, int M, int K, int mode, hipStream_t s) {
  dim3 g(M / 16), b(256);                       // all M are multiples of 16
  if (mode == 0)      gemm_wmma<0><<<g, b, 0, s>>>(X, Wp, bias, Y, nullptr, M, K);
  else if (mode == 1) gemm_wmma<1><<<g, b, 0, s>>>(X, Wp, bias, Y, nullptr, M, K);
  else                gemm_wmma<2><<<g, b, 0, s>>>(X, Wp, bias, nullptr, colsum, M, K);
}

static void run_edge_att(const float* hs, const float* hd, const float* a_s, const float* a_d,
                         const int* src, const int* dst, int nSrc, int nDst, int E,
                         float* asrc, float* adst, unsigned* mkey, float* segs,
                         float* alpha, float* out, hipStream_t s) {
  row_dot<<<(nSrc + 7) / 8, 256, 0, s>>>(hs, a_s, asrc, nSrc);
  row_dot<<<(nDst + 7) / 8, 256, 0, s>>>(hd, a_d, adst, nDst);
  hipMemsetAsync(mkey, 0, (size_t)nDst * sizeof(unsigned), s);   // key 0 < any finite
  hipMemsetAsync(segs, 0, (size_t)nDst * sizeof(float), s);
  hipMemsetAsync(out, 0, (size_t)nDst * HN * sizeof(float), s);
  edge_alpha_kernel<<<(E + 255) / 256, 256, 0, s>>>(asrc, adst, src, dst, alpha, mkey, E);
  edge_exp_kernel<<<(E + 255) / 256, 256, 0, s>>>(alpha, mkey, dst, segs, E);
  edge_scatter_kernel<<<E, HN, 0, s>>>(hs, alpha, segs, src, dst, out, E);
  relu_kernel<<<(int)(((size_t)nDst * HN + 255) / 256), 256, 0, s>>>(out, (size_t)nDst * HN);
}

extern "C" void kernel_launch(void* const* d_in, const int* in_sizes, int n_in,
                              void* d_out, int out_size, void* d_ws, size_t ws_size,
                              hipStream_t stream) {
  const float* user_emb     = (const float*)d_in[0];
  const float* visual_x     = (const float*)d_in[1];
  /* caption_x d_in[2] unused by the forward pass */
  const float* nutrient     = (const float*)d_in[3];
  const float* ingredient_x = (const float*)d_in[4];
  const float* cooking_x    = (const float*)d_in[5];
  const float* item_x       = (const float*)d_in[6];
  const float* nutrient_W   = (const float*)d_in[7];
  const float* nutrient_b   = (const float*)d_in[8];
  const float* ce1_W = (const float*)d_in[9];
  const float* ce1_b = (const float*)d_in[10];
  const float* ce2_W = (const float*)d_in[11];
  const float* ce2_b = (const float*)d_in[12];
  const float* proj_W1 = (const float*)d_in[13];   // [2,128,128]
  const float* proj_b1 = (const float*)d_in[14];   // [2,128]
  const float* att1    = (const float*)d_in[15];   // [2,2,128]
  /* k1_W/k1_b/q1 (d_in[16..18]) drop out: group() over one type is identity */
  const float* proj_W2 = (const float*)d_in[19];   // [6,128,128]
  const float* proj_b2 = (const float*)d_in[20];   // [6,128]
  const float* att2    = (const float*)d_in[21];   // [5,2,128]
  const float* k2_W    = (const float*)d_in[22];
  const float* k2_b    = (const float*)d_in[23];
  const float* q2      = (const float*)d_in[24];
  const float* bn_g    = (const float*)d_in[25];
  const float* bn_b    = (const float*)d_in[26];
  const int* user_ids = (const int*)d_in[27];
  const int* it_src   = (const int*)d_in[28];
  const int* it_dst   = (const int*)d_in[29];
  const int* ui_src   = (const int*)d_in[30];
  const int* ui_dst   = (const int*)d_in[31];

  // ---- workspace carve-up (floats), peak ~323 MB ----
  const size_t F_REC = (size_t)NUM_REC * HN;
  const size_t F_U   = (size_t)NUM_U * HN;
  const size_t F_ING = (size_t)NUM_ING * HN;
  float* w = (float*)d_ws;
  float* bufA = w;                 // nutr_h -> h_tst -> o_item0
  float* bufB = bufA + F_REC;      // z1 -> taste_x -> o_item1
  float* bufC = bufB + F_REC;      // intention_x -> o_item2
  float* bufD = bufC + F_REC;      // h1 -> item_out
  float* bufE = bufD + F_REC;      // o_item3
  float* bufU1 = bufE + F_REC;     // user_x -> o_user (user_out)
  float* bufU2 = bufU1 + F_U;      // h0
  float* bufG  = bufU2 + F_U;      // h_ing
  float* edgeA = bufG + F_ING;     // per-edge alpha/exp (max E_UI)
  float* p1    = edgeA + NEDGE_UI; // node scores (<= NUM_U)
  float* p2    = p1 + NUM_U;       // node scores (<= NUM_REC)
  unsigned* mkey = (unsigned*)(p2 + NUM_REC);  // segment max keys (<= NUM_REC)
  float* segs  = (float*)(mkey + NUM_REC);     // segment sums (<= NUM_REC)
  float* colsum = segs + NUM_REC;  // [4,128]
  float* attn   = colsum + 4 * HN; // [4] (padded to 8 for 32B alignment below)
  float* bnsum  = attn + 8;        // [128]
  float* bnsq   = bnsum + HN;      // [128]
  _Float16* wpack = (_Float16*)(bnsq + HN);    // 11 slots x 16384 f16, 32B aligned

  // ---- 0) pack all weight matrices into f16 WMMA B-fragment layout ----
  _Float16* WP_nutr = wpack + 0 * WSLOT;
  _Float16* WP_ce1  = wpack + 1 * WSLOT;
  _Float16* WP_ce2  = wpack + 2 * WSLOT;
  _Float16* WP_p10  = wpack + 3 * WSLOT;
  _Float16* WP_p11  = wpack + 4 * WSLOT;
  _Float16* WP_p20  = wpack + 5 * WSLOT;
  _Float16* WP_p21  = wpack + 6 * WSLOT;
  _Float16* WP_p23  = wpack + 7 * WSLOT;
  _Float16* WP_p24  = wpack + 8 * WSLOT;
  _Float16* WP_p25  = wpack + 9 * WSLOT;
  _Float16* WP_k2   = wpack + 10 * WSLOT;
  pack_w<<<(32 * HN + 255) / 256, 256, 0, stream>>>(nutrient_W, WP_nutr, 20);
  pack_w<<<64, 256, 0, stream>>>(ce1_W, WP_ce1, HN);
  pack_w<<<64, 256, 0, stream>>>(ce2_W, WP_ce2, HN);
  pack_w<<<64, 256, 0, stream>>>(proj_W1 + 0 * HN * HN, WP_p10, HN);
  pack_w<<<64, 256, 0, stream>>>(proj_W1 + 1 * HN * HN, WP_p11, HN);
  pack_w<<<64, 256, 0, stream>>>(proj_W2 + 0 * HN * HN, WP_p20, HN);
  pack_w<<<64, 256, 0, stream>>>(proj_W2 + 1 * HN * HN, WP_p21, HN);
  pack_w<<<64, 256, 0, stream>>>(proj_W2 + 3 * HN * HN, WP_p23, HN);
  pack_w<<<64, 256, 0, stream>>>(proj_W2 + 4 * HN * HN, WP_p24, HN);
  pack_w<<<64, 256, 0, stream>>>(proj_W2 + 5 * HN * HN, WP_p25, HN);
  pack_w<<<64, 256, 0, stream>>>(k2_W, WP_k2, HN);

  // ---- 1) intention: nutrient linear -> MLP -> L2 normalize ----
  launch_gemm(nutrient, WP_nutr, nutrient_b, bufA, nullptr, NUM_REC, 20, 0, stream);
  launch_gemm(bufA, WP_ce1, ce1_b, bufB, nullptr, NUM_REC, HN, 1, stream);
  launch_gemm(bufB, WP_ce2, ce2_b, bufC, nullptr, NUM_REC, HN, 0, stream);
  l2_normalize<<<(NUM_REC + 7) / 8, 256, 0, stream>>>(bufC, NUM_REC);   // bufC = intention_x

  // ---- 2) user embedding lookup with max_norm=1 renorm ----
  gather_renorm<<<(NUM_U + 7) / 8, 256, 0, stream>>>(user_emb, user_ids, bufU1, NUM_U);

  // ---- 3) ing_to_recipe HANConv (single edge type => group() identity) ----
  launch_gemm(ingredient_x, WP_p10, proj_b1 + 0 * HN, bufG, nullptr, NUM_ING, HN, 0, stream);
  launch_gemm(cooking_x,    WP_p11, proj_b1 + 1 * HN, bufA, nullptr, NUM_REC, HN, 0, stream);
  run_edge_att(bufG, bufA, att1 + 0, att1 + HN, it_src, it_dst,
               NUM_ING, NUM_REC, NEDGE_IT, p1, p2, mkey, segs, edgeA, bufB, stream); // bufB = taste_x

  // ---- 4) main HANConv projections (identity-edge types fold to relu(h)) ----
  launch_gemm(bufB, WP_p23, proj_b2 + 3 * HN, bufA, nullptr, NUM_REC, HN, 1, stream); // o_item0
  launch_gemm(bufC, WP_p24, proj_b2 + 4 * HN, bufB, nullptr, NUM_REC, HN, 1, stream); // o_item1
  launch_gemm(visual_x, WP_p25, proj_b2 + 5 * HN, bufC, nullptr, NUM_REC, HN, 1, stream); // o_item2
  launch_gemm(bufU1, WP_p20, proj_b2 + 0 * HN, bufU2, nullptr, NUM_U, HN, 0, stream);  // h0
  launch_gemm(item_x, WP_p21, proj_b2 + 1 * HN, bufD, nullptr, NUM_REC, HN, 0, stream); // h1

  // ---- 5) user->item attention (o_item3) and item->user attention (o_user) ----
  run_edge_att(bufU2, bufD, att2 + 3 * 2 * HN, att2 + 3 * 2 * HN + HN, ui_src, ui_dst,
               NUM_U, NUM_REC, NEDGE_UI, p1, p2, mkey, segs, edgeA, bufE, stream);   // bufE = o_item3
  run_edge_att(bufD, bufU2, att2 + 4 * 2 * HN, att2 + 4 * 2 * HN + HN, ui_dst, ui_src,
               NUM_REC, NUM_U, NEDGE_UI, p2, p1, mkey, segs, edgeA, bufU1, stream);  // bufU1 = user_out

  // ---- 6) semantic attention over the 4 item edge types ----
  hipMemsetAsync(colsum, 0, 4 * HN * sizeof(float), stream);
  launch_gemm(bufA, WP_k2, k2_b, nullptr, colsum + 0 * HN, NUM_REC, HN, 2, stream);
  launch_gemm(bufB, WP_k2, k2_b, nullptr, colsum + 1 * HN, NUM_REC, HN, 2, stream);
  launch_gemm(bufC, WP_k2, k2_b, nullptr, colsum + 2 * HN, NUM_REC, HN, 2, stream);
  launch_gemm(bufE, WP_k2, k2_b, nullptr, colsum + 3 * HN, NUM_REC, HN, 2, stream);
  group_softmax<<<1, 128, 0, stream>>>(colsum, q2, attn, 1.0f / (float)NUM_REC);
  combine4<<<(int)((F_REC + 255) / 256), 256, 0, stream>>>(bufA, bufB, bufC, bufE, attn, bufD, F_REC); // bufD = item_out

  // ---- 7) BatchNorm (batch stats) per node type, concat into d_out ----
  float* out_f = (float*)d_out;
  hipMemsetAsync(bnsum, 0, HN * sizeof(float), stream);
  hipMemsetAsync(bnsq, 0, HN * sizeof(float), stream);
  col_stats<<<512, 128, 0, stream>>>(bufU1, NUM_U, bnsum, bnsq);
  bn_apply<<<512, 128, 0, stream>>>(bufU1, bnsum, bnsq, bn_g, bn_b, out_f, NUM_U);

  hipMemsetAsync(bnsum, 0, HN * sizeof(float), stream);
  hipMemsetAsync(bnsq, 0, HN * sizeof(float), stream);
  col_stats<<<512, 128, 0, stream>>>(bufD, NUM_REC, bnsum, bnsq);
  bn_apply<<<512, 128, 0, stream>>>(bufD, bnsum, bnsq, bn_g, bn_b, out_f + F_U, NUM_REC);
}
